// SO2Convolution_3470333575341
// MI455X (gfx1250) — matrix operand order repaired and checked by reading, not compiled, not run on hardware
//
#include <hip/hip_runtime.h>
#include <cstdint>

// SO2Convolution for MI455X (gfx1250, wave32, WMMA).
//
// out(E, 29*128) = blockdiag-GEMM of x(E, 29*128) against 3 weight blocks:
//   seg0: K=N=896   (w_m0 + bias)
//   seg1: K=N=1536  (B' = [[Wr, Wi], [-Wi, Wr]] from w_m1)
//   seg2: K=N=1280  (B' from w_m2)
// f16 inputs via v_cvt_pk_rtz, f32 accumulation via v_wmma_f32_16x16x32_f16.

#define E_TOT     60000
#define ROWSTRIDE 3712            // 29*128 floats, both x and out

typedef __attribute__((ext_vector_type(16))) _Float16 v16h;
typedef __attribute__((ext_vector_type(2)))  __fp16   fp16x2;  // cvt_pkrtz return type
typedef __attribute__((ext_vector_type(8)))  float    v8f;

#define PACK_OFF0 0
#define PACK_OFF1 (896*896)
#define PACK_OFF2 (896*896 + 1536*1536)
#define PACK_TOTAL (896*896 + 1536*1536 + 1280*1280)   // 4,800,512 halfs = 9.2 MB

// ---- Weight pre-pack into WMMA B-fragment order -----------------------------
// Packed layout (per segment): frag (nfrag, kt) -> 32 lanes x 16 f16 (1 KB).
//   offset = ((nfrag*numKt + kt)*32 + lane)*16 + e
//   lane<16 : element e = B[kt*32 + (e&7) + (e>>3)*16][nfrag*16 + lane]
//   lane>=16: element e = B[kt*32 + 8 + (e&7) + (e>>3)*16][nfrag*16 + lane-16]

__global__ void pack_m0(const float* __restrict__ w, _Float16* __restrict__ dst) {
  int t = blockIdx.x * blockDim.x + threadIdx.x;
  if (t >= 896 * 896) return;
  int e    = t & 15;
  int lane = (t >> 4) & 31;
  int rest = t >> 9;
  const int numKt = 28;                  // 896/32
  int kt    = rest % numKt;
  int nfrag = rest / numKt;
  int K = kt * 32 + (lane >> 4) * 8 + (e & 7) + (e >> 3) * 16;
  int N = nfrag * 16 + (lane & 15);
  dst[t] = (_Float16)w[K * 896 + N];
}

// B'[k][n] for the SO(2) complex structure, w is (Kh, 2*Kh):
//   k <  Kh            : B' =  w[k][n]                (covers both Wr and Wi cols)
//   k >= Kh, n <  Kh   : B' = -w[k-Kh][n+Kh]          (-Wi)
//   k >= Kh, n >= Kh   : B' =  w[k-Kh][n-Kh]          ( Wr)
__global__ void pack_so2(const float* __restrict__ w, _Float16* __restrict__ dst,
                         int Kh, int numKt) {
  int t = blockIdx.x * blockDim.x + threadIdx.x;
  int twoKh = Kh * 2;
  if (t >= twoKh * twoKh) return;
  int e    = t & 15;
  int lane = (t >> 4) & 31;
  int rest = t >> 9;
  int kt    = rest % numKt;
  int nfrag = rest / numKt;
  int K = kt * 32 + (lane >> 4) * 8 + (e & 7) + (e >> 3) * 16;
  int N = nfrag * 16 + (lane & 15);
  float v;
  if (K < Kh)      v =  w[K * twoKh + N];
  else if (N < Kh) v = -w[(K - Kh) * twoKh + (N + Kh)];
  else             v =  w[(K - Kh) * twoKh + (N - Kh)];
  dst[t] = (_Float16)v;
}

// ---- Main GEMM --------------------------------------------------------------
// Block = 256 threads = 8 waves as 2(M) x 4(N). Wave tile = 64x32.
// Each workgroup: 128 edges x full N of its segment (A slab stays hot in L0/L2).

template <bool GUARD>
__device__ __forceinline__ void store_tile(
    float* __restrict__ out, const v8f acc[4][2], int seg,
    const float* __restrict__ b_m0, int m0, int wave_m, int wave_n,
    int r, int half, int nt, int colBase) {
  #pragma unroll
  for (int nf = 0; nf < 2; ++nf) {
    int ncol = nt * 128 + wave_n * 32 + nf * 16 + r;
    float bias = (seg == 0) ? b_m0[ncol] : 0.0f;
    int col = colBase + ncol;
    #pragma unroll
    for (int f = 0; f < 4; ++f) {
      int rowBase = m0 + wave_m * 64 + f * 16 + half * 8;
      #pragma unroll
      for (int e = 0; e < 8; ++e) {
        int row = rowBase + e;
        if (!GUARD || row < E_TOT)
          out[(size_t)row * ROWSTRIDE + col] = acc[f][nf][e] + bias;
      }
    }
  }
}

__global__ __launch_bounds__(256) void so2_wmma_gemm(
    const float* __restrict__ x, const float* __restrict__ b_m0,
    const _Float16* __restrict__ wpack, float* __restrict__ out) {
  const int seg = blockIdx.y;
  int K, colBase, packOff;
  if (seg == 0)      { K = 896;  colBase = 0;    packOff = PACK_OFF0; }
  else if (seg == 1) { K = 1536; colBase = 896;  packOff = PACK_OFF1; }
  else               { K = 1280; colBase = 2432; packOff = PACK_OFF2; }
  const int numKt  = K >> 5;   // K tiles of 32
  const int ntiles = K >> 7;   // N == K, N tiles of 128

  const int tid    = threadIdx.x;
  const int lane   = tid & 31;
  const int wid    = tid >> 5;
  const int wave_m = wid >> 2;     // 0..1 -> 64-row half
  const int wave_n = wid & 3;      // 0..3 -> 32-col quarter
  const int r      = lane & 15;
  const int half   = lane >> 4;

  const int m0 = blockIdx.x * 128;
  const bool fullTile = (m0 + 128 <= E_TOT);   // workgroup-uniform

  // Clamped per-fragment A row pointers (already offset by half*8 K-phase).
  const float* arow[4];
  #pragma unroll
  for (int f = 0; f < 4; ++f) {
    int row = m0 + wave_m * 64 + f * 16 + r;
    if (row >= E_TOT) row = E_TOT - 1;       // tail clamp; stores are guarded
    arow[f] = x + (size_t)row * ROWSTRIDE + colBase + half * 8;
  }

  const v8f vzero = {0.f, 0.f, 0.f, 0.f, 0.f, 0.f, 0.f, 0.f};

  for (int nt = 0; nt < ntiles; ++nt) {
    v8f acc[4][2];
    #pragma unroll
    for (int f = 0; f < 4; ++f) { acc[f][0] = vzero; acc[f][1] = vzero; }

    const _Float16* bbase0 = wpack + packOff +
        ((size_t)((nt * 8 + wave_n * 2 + 0) * numKt) * 32 + lane) * 16;
    const _Float16* bbase1 = wpack + packOff +
        ((size_t)((nt * 8 + wave_n * 2 + 1) * numKt) * 32 + lane) * 16;

    for (int kt = 0; kt < numKt; ++kt) {
      const int k = kt << 5;
      v16h a[4];
      #pragma unroll
      for (int f = 0; f < 4; ++f) {
        const float4* p = (const float4*)(arow[f] + k);
        float4 c0 = p[0];      // K = k + half*8 + {0..3}
        float4 c1 = p[1];      // K = k + half*8 + {4..7}
        float4 c2 = p[4];      // K = k + 16 + half*8 + {0..3}
        float4 c3 = p[5];      // K = k + 16 + half*8 + {4..7}
        union { v16h v; fp16x2 h[8]; } af;
        af.h[0] = __builtin_amdgcn_cvt_pkrtz(c0.x, c0.y);
        af.h[1] = __builtin_amdgcn_cvt_pkrtz(c0.z, c0.w);
        af.h[2] = __builtin_amdgcn_cvt_pkrtz(c1.x, c1.y);
        af.h[3] = __builtin_amdgcn_cvt_pkrtz(c1.z, c1.w);
        af.h[4] = __builtin_amdgcn_cvt_pkrtz(c2.x, c2.y);
        af.h[5] = __builtin_amdgcn_cvt_pkrtz(c2.z, c2.w);
        af.h[6] = __builtin_amdgcn_cvt_pkrtz(c3.x, c3.y);
        af.h[7] = __builtin_amdgcn_cvt_pkrtz(c3.z, c3.w);
        a[f] = af.v;
      }
      v16h b0 = *(const v16h*)(bbase0 + (size_t)kt * 512);
      v16h b1 = *(const v16h*)(bbase1 + (size_t)kt * 512);
      #pragma unroll
      for (int f = 0; f < 4; ++f) {
        acc[f][0] = __builtin_amdgcn_wmma_f32_16x16x32_f16(
            false, a[f], false, b0, (short)0, acc[f][0], false, false);
        acc[f][1] = __builtin_amdgcn_wmma_f32_16x16x32_f16(
            false, a[f], false, b1, (short)0, acc[f][1], false, false);
      }
    }

    // Store: C/D layout -> element e of lane: row = base + half*8 + e, col = lane%16.
    // Uniform branch: only the single tail M-tile takes the guarded path.
    if (fullTile)
      store_tile<false>(out, acc, seg, b_m0, m0, wave_m, wave_n, r, half, nt, colBase);
    else
      store_tile<true >(out, acc, seg, b_m0, m0, wave_m, wave_n, r, half, nt, colBase);
  }
}

extern "C" void kernel_launch(void* const* d_in, const int* in_sizes, int n_in,
                              void* d_out, int out_size, void* d_ws, size_t ws_size,
                              hipStream_t stream) {
  const float* x    = (const float*)d_in[0];
  // d_in[1] = x_edge : unused by the reference computation
  const float* w_m0 = (const float*)d_in[2];
  const float* b_m0 = (const float*)d_in[3];
  const float* w_m1 = (const float*)d_in[4];
  const float* w_m2 = (const float*)d_in[5];
  float* out = (float*)d_out;
  _Float16* wpack = (_Float16*)d_ws;   // needs 9,601,024 bytes of workspace

  pack_m0 <<<(896*896   + 255) / 256, 256, 0, stream>>>(w_m0, wpack + PACK_OFF0);
  pack_so2<<<(1536*1536 + 255) / 256, 256, 0, stream>>>(w_m1, wpack + PACK_OFF1, 768, 48);
  pack_so2<<<(1280*1280 + 255) / 256, 256, 0, stream>>>(w_m2, wpack + PACK_OFF2, 640, 40);

  dim3 grid((E_TOT + 127) / 128, 3);
  so2_wmma_gemm<<<grid, 256, 0, stream>>>(x, b_m0, wpack, out);
}
